// EncodingCosine_14053132992763
// MI455X (gfx1250) — matrix operand order repaired and checked by reading, not compiled, not run on hardware
//
#include <hip/hip_runtime.h>
#include <hip/hip_bf16.h>
#include <math.h>

// ---------------------------------------------------------------------------
// EncodingCosine fused kernel for MI455X (gfx1250, wave32, WMMA bf16)
//   GEMM1 (cosine logits) + softmax + GEMM2 (aggregation), one pass over X.
// ---------------------------------------------------------------------------

typedef __bf16 bf16;
typedef bf16  v16bf __attribute__((ext_vector_type(16)));
typedef bf16  v8bf  __attribute__((ext_vector_type(8)));
typedef float v8f   __attribute__((ext_vector_type(8)));

#define Bn      32
#define Dn      256
#define Nn      4096
#define Kn      64
#define NT      128      // n-rows staged per iteration
#define NSPLIT  8        // workgroups per batch
#define THREADS 256      // 8 wave32
#define EPSF    1e-8f

// LDS layout (dynamic shared memory, ~153 KB -> 2 WGs / 320KB WGP)
#define XS_STR (Dn + 8)   // Xs  [NT][XS_STR]  bf16, n-major (GEMM1 A operand)
#define XT_STR (NT + 8)   // XsT [Dn][XT_STR]  bf16, d-major (GEMM2 B operand)
#define AT_STR (NT + 8)   // At  [Kn][AT_STR]  bf16, k-major (GEMM2 A operand)
#define OFF_XS 0
#define OFF_XT (OFF_XS + NT * XS_STR * 2)
#define OFF_AT (OFF_XT + Dn * XT_STR * 2)
#define OFF_RN (OFF_AT + Kn * AT_STR * 2)     // float rnorm[NT]
#define OFF_S  (OFF_RN + NT * 4)              // float sArr[Kn]
#define OFF_PS (OFF_S  + Kn * 4)              // float partsq[THREADS]
#define SMEM_TOTAL (OFF_PS + THREADS * 4)

// A-operand fragment (16-bit A 16x32 layout): element e<8 -> K=d0+e(+8*hi),
// e>=8 -> K=d0+16+(e-8)(+8*hi). Caller passes p = &row[d0 + 8*hi];
// chunks live at p[0..7] and p[16..23]. 16B-aligned LDS/global loads.
static __device__ inline v16bf load_frag_a(const bf16* p) {
  v8bf lo = *(const v8bf*)(p);
  v8bf hh = *(const v8bf*)(p + 16);
  v16bf r;
#pragma unroll
  for (int e = 0; e < 8; ++e) { r[e] = lo[e]; r[e + 8] = hh[e]; }
  return r;
}

// B-operand fragment (contiguous K order, not AMD's table order): lane half selects K-half,
// elements are 16 consecutive K values. Caller passes p = &col[d0 + 16*hi].
static __device__ inline v16bf load_frag_b(const bf16* p) {
  v8bf lo = *(const v8bf*)(p);
  v8bf hh = *(const v8bf*)(p + 8);
  v16bf r;
#pragma unroll
  for (int e = 0; e < 8; ++e) { r[e] = lo[e]; r[e + 8] = hh[e]; }
  return r;
}

// --------------------------- codeword prep --------------------------------
// Normalize codewords (K x D, f32) -> bf16 in workspace.
__global__ __launch_bounds__(Dn) void prep_codewords(const float* __restrict__ C,
                                                     bf16* __restrict__ Cn) {
  __shared__ float red[Dn / 32];
  const int k = blockIdx.x;
  const int t = threadIdx.x;
  float x = C[k * Dn + t];
  float s = x * x;
#pragma unroll
  for (int m = 1; m < 32; m <<= 1) s += __shfl_xor(s, m, 32);
  if ((t & 31) == 0) red[t >> 5] = s;
  __syncthreads();
  float tot = 0.f;
#pragma unroll
  for (int i = 0; i < Dn / 32; ++i) tot += red[i];
  float rn = 1.0f / fmaxf(sqrtf(tot), EPSF);
  Cn[k * Dn + t] = (bf16)(x * rn);
}

// ----------------------------- main kernel --------------------------------
__global__ __launch_bounds__(THREADS) void encode_fused(
    const float* __restrict__ Xg, const float* __restrict__ Cg,
    const bf16* __restrict__ Cn, float* __restrict__ out) {
  extern __shared__ char smem[];
  bf16*  Xs  = (bf16*)(smem + OFF_XS);
  bf16*  XsT = (bf16*)(smem + OFF_XT);
  bf16*  At  = (bf16*)(smem + OFF_AT);
  float* rno = (float*)(smem + OFF_RN);
  float* sAr = (float*)(smem + OFF_S);
  float* psq = (float*)(smem + OFF_PS);

  const int t    = threadIdx.x;
  const int lane = t & 31;
  const int w    = t >> 5;        // wave 0..7
  const int ln   = lane & 15;
  const int hi   = lane >> 4;     // half-wave 0/1
  const int b    = blockIdx.x / NSPLIT;
  const int spl  = blockIdx.x % NSPLIT;
  const int n0   = spl * (Nn / NSPLIT);   // 512-wide n chunk
  const int m0   = w * 16;                // GEMM1: 16-row n strip per wave

  if (t < Kn) sAr[t] = 0.0f;

  // GEMM2 accumulators: wave owns k-strip [ks,ks+16) x d-strip [dbase,dbase+128)
  const int ks    = (w & 3) * 16;
  const int dbase = (w >> 2) * 128;
  v8f E[8];
#pragma unroll
  for (int i = 0; i < 8; ++i)
#pragma unroll
    for (int r = 0; r < 8; ++r) E[i][r] = 0.0f;

  const int iters = (Nn / NSPLIT) / NT;   // 4
  for (int it = 0; it < iters; ++it) {
    const int nb = n0 + it * NT;
    __syncthreads();   // LDS reuse barrier (also orders sAr init on it==0)

    // ---- stage X tile: f32 global -> bf16 LDS (both majors) + row norms ----
    float ssq = 0.f;
    const int j  = t & 127;      // n offset within tile
    const int dh = t >> 7;       // 0/1
    if (it + 1 < iters)          // prefetch a line of the next tile per d-row
      __builtin_prefetch(&Xg[((size_t)(b * Dn + t)) * Nn + nb + NT], 0, 1);
#pragma unroll 4
    for (int iter = 0; iter < 128; ++iter) {
      const int d = iter * 2 + dh;
      float x = Xg[((size_t)(b * Dn + d)) * Nn + nb + j];
      bf16 h = (bf16)x;
      Xs[j * XS_STR + d]  = h;
      XsT[d * XT_STR + j] = h;
      ssq += x * x;
    }
    psq[t] = ssq;
    __syncthreads();
    if (t < NT) {
      float s2 = psq[t] + psq[t + NT];
      rno[t] = 1.0f / fmaxf(sqrtf(s2), EPSF);
    }
    __syncthreads();

    // ---- GEMM1: Lraw = Xraw . CnormT  (scale by 1/||x|| in epilogue) ----
    v8f acc[4];
#pragma unroll
    for (int kt = 0; kt < 4; ++kt)
#pragma unroll
      for (int r = 0; r < 8; ++r) acc[kt][r] = 0.0f;

#pragma unroll
    for (int dstep = 0; dstep < 8; ++dstep) {
      const int d0 = dstep * 32;
      v16bf a = load_frag_a(&Xs[(m0 + ln) * XS_STR + d0 + 8 * hi]);
#pragma unroll
      for (int kt = 0; kt < 4; ++kt) {
        v16bf bb = load_frag_a(&Cn[(kt * 16 + ln) * Dn + d0 + 8 * hi]);
        acc[kt] = __builtin_amdgcn_wmma_f32_16x16x32_bf16(
            false, a, false, bb, (short)0, acc[kt], false, false);
      }
    }

    // ---- softmax over k (row = n): D-tile layout row m = m0+r+8*hi ----
    float rn8[8], rmax[8], rsum[8];
#pragma unroll
    for (int r = 0; r < 8; ++r) rn8[r] = rno[m0 + r + 8 * hi];
#pragma unroll
    for (int kt = 0; kt < 4; ++kt)
#pragma unroll
      for (int r = 0; r < 8; ++r) acc[kt][r] *= rn8[r];
#pragma unroll
    for (int r = 0; r < 8; ++r)
      rmax[r] = fmaxf(fmaxf(acc[0][r], acc[1][r]), fmaxf(acc[2][r], acc[3][r]));
#pragma unroll
    for (int m = 1; m < 16; m <<= 1)
#pragma unroll
      for (int r = 0; r < 8; ++r) rmax[r] = fmaxf(rmax[r], __shfl_xor(rmax[r], m, 32));
#pragma unroll
    for (int r = 0; r < 8; ++r) rsum[r] = 0.0f;
#pragma unroll
    for (int kt = 0; kt < 4; ++kt)
#pragma unroll
      for (int r = 0; r < 8; ++r) {
        acc[kt][r] = __expf(acc[kt][r] - rmax[r]);
        rsum[r] += acc[kt][r];
      }
#pragma unroll
    for (int m = 1; m < 16; m <<= 1)
#pragma unroll
      for (int r = 0; r < 8; ++r) rsum[r] += __shfl_xor(rsum[r], m, 32);
#pragma unroll
    for (int r = 0; r < 8; ++r) rsum[r] = 1.0f / rsum[r];

    // store A (k-major) + accumulate column sums s[k]
#pragma unroll
    for (int kt = 0; kt < 4; ++kt) {
      float sloc = 0.0f;
#pragma unroll
      for (int r = 0; r < 8; ++r) {
        float Av = acc[kt][r] * rsum[r];
        At[(kt * 16 + ln) * AT_STR + m0 + r + 8 * hi] = (bf16)Av;
        sloc += Av;
      }
      atomicAdd(&sAr[kt * 16 + ln], sloc);   // ds_add_f32
    }
    __syncthreads();

    // ---- GEMM2: E[k,d] += sum_n A[k,n] * Xraw[n,d] ----
#pragma unroll
    for (int ns = 0; ns < 4; ++ns) {
      const int nn = ns * 32;
      v16bf a = load_frag_a(&At[(ks + ln) * AT_STR + nn + 8 * hi]);
#pragma unroll
      for (int dt = 0; dt < 8; ++dt) {
        v16bf bb = load_frag_a(&XsT[(dbase + dt * 16 + ln) * XT_STR + nn + 8 * hi]);
        E[dt] = __builtin_amdgcn_wmma_f32_16x16x32_bf16(
            false, a, false, bb, (short)0, E[dt], false, false);
      }
    }
  }
  __syncthreads();

  // ---- epilogue: E_partial - s_partial[k]*C[k,d], atomic add to output ----
#pragma unroll
  for (int r = 0; r < 8; ++r) {
    const int k = ks + r + 8 * hi;
    const float sk = sAr[k];
#pragma unroll
    for (int dt = 0; dt < 8; ++dt) {
      const int d = dbase + dt * 16 + ln;
      float val = E[dt][r] - sk * Cg[k * Dn + d];
      float* dst = &out[((size_t)b * Kn + k) * Dn + d];
      __hip_atomic_fetch_add(dst, val, __ATOMIC_RELAXED, __HIP_MEMORY_SCOPE_AGENT);
    }
  }
}

// ------------------------------- launcher ---------------------------------
extern "C" void kernel_launch(void* const* d_in, const int* in_sizes, int n_in,
                              void* d_out, int out_size, void* d_ws, size_t ws_size,
                              hipStream_t stream) {
  const float* X  = (const float*)d_in[0];     // (32, 256, 64, 64) f32
  const float* C  = (const float*)d_in[1];     // (64, 256) f32
  float* out      = (float*)d_out;             // (32, 64, 256) f32
  bf16* Cn        = (bf16*)d_ws;               // 32 KB normalized codewords

  hipMemsetAsync(d_out, 0, (size_t)out_size * sizeof(float), stream);
  prep_codewords<<<Kn, Dn, 0, stream>>>(C, Cn);

  hipFuncSetAttribute((const void*)encode_fused,
                      hipFuncAttributeMaxDynamicSharedMemorySize, SMEM_TOTAL);
  encode_fused<<<Bn * NSPLIT, THREADS, SMEM_TOTAL, stream>>>(X, C, Cn, out);
}
